// DeepEZGAT_82033875354137
// MI455X (gfx1250) — compile-verified, matrix operand order checked by reading, not run on hardware
//
#include <hip/hip_runtime.h>
#include <cstdint>
#include <cstddef>

// ---------------------------------------------------------------------------
// CDNA5 (gfx1250) wave32 WMMA types and helpers
// ---------------------------------------------------------------------------
typedef __attribute__((ext_vector_type(16))) __bf16 v16bf;
typedef __attribute__((ext_vector_type(8)))  float  v8f;

__device__ __forceinline__ v8f wmma_bf16(v16bf a, v16bf b, v8f c) {
  // D = A(16x32 bf16) * B(32x16 bf16) + C(16x16 f32)
  return __builtin_amdgcn_wmma_f32_16x16x32_bf16(
      /*neg_a=*/false, a, /*neg_b=*/false, b,
      /*c_mod=*/(short)0, c, /*reuse_a=*/false, /*reuse_b=*/false);
}

// Async global->LDS copy (CDNA5, tracked with ASYNCcnt).
__device__ __forceinline__ void async_ld_b128(unsigned lds_off, const void* gptr) {
  asm volatile("global_load_async_to_lds_b128 %0, %1, off"
               :: "v"(lds_off), "v"(gptr)
               : "memory");
}
__device__ __forceinline__ void wait_async0() {
  asm volatile("s_wait_asynccnt 0" ::: "memory");
}

__device__ __forceinline__ float leaky(float x) {
  return (x >= 0.0f) ? x : 0.1f * x;
}

static constexpr int N_NODES = 6000;
static constexpr int NW      = 188;     // ceil(6000/32) mask words per row
static constexpr int NWPAD   = NW * 32; // 6016

// B-fragment swizzle: element (kk, n-within-tile) -> [lane][u] slot so a lane
// can load its whole v16bf with two ds_load_b128.
//   lane = 16*((kk>>3)&1) + n ;  u = (kk&7) + (kk>=16 ? 8 : 0)
__device__ __forceinline__ int bfrag_lane(int kk, int n) {
  return ((kk >> 3) & 1) * 16 + n;
}
__device__ __forceinline__ int bfrag_u(int kk) {
  return (kk & 7) + ((kk >= 16) ? 8 : 0);
}

// ---------------------------------------------------------------------------
// Kernel 1: adjacency -> packed edge bitmask (read adj once, 144MB -> 4.5MB,
// L2-resident for all 5 attention passes)
// ---------------------------------------------------------------------------
__global__ __launch_bounds__(256) void build_mask(const float* __restrict__ adj,
                                                  uint32_t* __restrict__ mask) {
  long long gid = (long long)blockIdx.x * 256 + threadIdx.x;
  int row = (int)(gid / NWPAD);
  int col = (int)(gid % NWPAD);
  bool pred = (col < N_NODES) && (adj[(size_t)row * N_NODES + col] > 0.0f);
  unsigned long long b = __ballot(pred);          // wave32 -> low 32 bits
  if ((threadIdx.x & 31) == 0)
    mask[(size_t)row * NW + (col >> 5)] = (uint32_t)b;
}

// ---------------------------------------------------------------------------
// Kernel 2: generic GEMM C[M,Nc] = A[M,K] @ B[K,Nc], f32 I/O, bf16 WMMA core.
//   bmode==1 : B is GAT w1 layout [4][K][30] flattened, col = h*30 + o
//   A tile (128 x 32 f32) staged into LDS with async b128 copies.
//   B tile staged as bf16 pre-swizzled into WMMA fragment order.
// ---------------------------------------------------------------------------
template<int NT>
__global__ __launch_bounds__(256) void gemm_bf16(
    const float* __restrict__ A, int lda,
    const float* __restrict__ B, int ldb, int bmode,
    float* __restrict__ C, int ldc,
    int M, int Ncols, int K)
{
  constexpr int BN = NT * 16;
  constexpr int AS = 36;            // LDS row stride (floats), 144B (16B aligned)
  __shared__ float sA[128 * AS];
  __shared__ alignas(32) __bf16 sBf[NT * 32 * 16];   // [tile][lane][u]

  const int tid  = threadIdx.x;
  const int wave = tid >> 5;
  const int lane = tid & 31;
  const int rowBase = blockIdx.x * 128;

  v8f acc[NT] = {};

  const int nk = (K + 31) / 32;
  for (int kt = 0; kt < nk; ++kt) {
    const int k0   = kt * 32;
    const int krem = (K - k0 < 32) ? (K - k0) : 32;

    // ---- stage A tile: 128 rows x 32 f32, 1024 b128 transfers, 4 per thread
    for (int it = 0; it < 4; ++it) {
      int xfer = tid + it * 256;       // 0..1023
      int r    = xfer >> 3;            // row in tile
      int cq   = (xfer & 7) * 4;       // float offset inside the 32-wide tile
      int gr   = rowBase + r; if (gr >= M) gr = M - 1;
      int kq   = k0 + cq;   if (kq + 4 > K) kq = K - 4;   // K multiple of 4
      const float* gp = A + (size_t)gr * lda + kq;
      unsigned lo = (unsigned)(uintptr_t)&sA[r * AS + cq];
      async_ld_b128(lo, gp);
    }
    wait_async0();

    // ---- stage B tile: bf16, fragment-order swizzle (shared by all 8 waves)
    for (int idx = tid; idx < 32 * BN; idx += 256) {
      int kk = idx / BN, c = idx % BN;
      int gk = k0 + kk;
      float v = 0.0f;
      if (gk < K && c < Ncols) {
        if (bmode == 1)
          v = B[(size_t)(c / 30) * (size_t)K * 30 + (size_t)gk * 30 + (c % 30)];
        else
          v = B[(size_t)gk * ldb + c];
      }
      int t = c >> 4, n = c & 15;
      sBf[(t * 32 + bfrag_lane(kk, n)) * 16 + bfrag_u(kk)] = (__bf16)v;
    }
    __syncthreads();

    if (krem < 32) {                   // zero-fill K padding in the A tile
      int span = 32 - krem;
      for (int idx = tid; idx < 128 * span; idx += 256) {
        int r = idx / span, kk = krem + idx % span;
        sA[r * AS + kk] = 0.0f;
      }
      __syncthreads();
    }

    // ---- build fragments + WMMA
    const int m    = lane & 15;
    const int base = (lane < 16) ? 0 : 8;   // A: K-half per lane group
    v16bf a;
    const float* ar = &sA[(wave * 16 + m) * AS];
    #pragma unroll
    for (int t = 0; t < 8; ++t) {
      a[t]     = (__bf16)ar[base + t];
      a[8 + t] = (__bf16)ar[base + 16 + t];
    }
    #pragma unroll
    for (int t = 0; t < NT; ++t) {
      v16bf b = *((const v16bf*)&sBf[(t * 32 + lane) * 16]);
      acc[t] = wmma_bf16(a, b, acc[t]);
    }
    __syncthreads();
  }

  // ---- store C (layout: lane n = lane&15, rows = 8*(lane>=16)+v)
  const int r0 = wave * 16 + 8 * (lane >> 4);
  #pragma unroll
  for (int t = 0; t < NT; ++t) {
    int col = t * 16 + (lane & 15);
    if (col < Ncols) {
      #pragma unroll
      for (int v = 0; v < 8; ++v) {
        int gr = rowBase + r0 + v;
        if (gr < M) C[(size_t)gr * ldc + col] = acc[t][v];
      }
    }
  }
}

// ---------------------------------------------------------------------------
// Kernel 3: attention coefficients f1/f2 = wh . a_src / a_dst   (tiny)
// ---------------------------------------------------------------------------
__global__ void attn_coeff(const float* __restrict__ wh, int ldw, int O, int heads,
                           const float* __restrict__ asrc, const float* __restrict__ adst,
                           float* __restrict__ f1, float* __restrict__ f2)
{
  int gid = blockIdx.x * blockDim.x + threadIdx.x;
  if (gid >= heads * N_NODES) return;
  int h = gid / N_NODES, n = gid % N_NODES;
  const float* r = wh + (size_t)n * ldw + h * O;
  float s1 = 0.0f, s2 = 0.0f;
  for (int o = 0; o < O; ++o) {
    float v = r[o];
    s1 += v * asrc[h * O + o];
    s2 += v * adst[h * O + o];
  }
  f1[gid] = s1;
  f2[gid] = s2;
}

// ---------------------------------------------------------------------------
// Kernel 4: fused masked-softmax attention + aggregate (flash-style).
// One wave owns 16 rows of one head; online softmax over 6000 cols in blocks
// of 32. The wh B tile is identical for all 8 waves of a block (same head),
// so it is staged cooperatively into LDS as bf16 in fragment order once per
// j-step; exp-weights become the bf16 A fragment.
// Epilogue: divide by row sum, add bias, leaky, store.
// ---------------------------------------------------------------------------
template<int NT>
__global__ __launch_bounds__(256) void attn_fused(
    const uint32_t* __restrict__ maskbits,
    const float* __restrict__ wh, int ldw, int colStride, int nValid,
    const float* __restrict__ f1, const float* __restrict__ f2,
    const float* __restrict__ bias,
    float* __restrict__ out, int ldo,
    int blocksPerHead, int numRB)
{
  constexpr int BN = NT * 16;
  __shared__ float f2s[N_NODES];
  __shared__ alignas(32) __bf16 sBf[NT * 32 * 16];   // [tile][lane][u]

  const int tid  = threadIdx.x;
  const int wave = tid >> 5;
  const int lane = tid & 31;
  const int head = blockIdx.x / blocksPerHead;
  int rb = (blockIdx.x % blocksPerHead) * 8 + wave;
  const bool active = (rb < numRB);
  if (!active) rb = numRB - 1;       // idle waves compute on a valid row block
  const int colOff = head * colStride;

  const float* f2h = f2 + (size_t)head * N_NODES;
  for (int i = tid; i < N_NODES; i += 256) f2s[i] = f2h[i];
  __syncthreads();

  const int i0   = rb * 16;
  const int m    = lane & 15;
  const int base = (lane < 16) ? 0 : 8;
  const int row  = i0 + m;
  const float f1v = f1[(size_t)head * N_NODES + row];

  float mrun = -INFINITY, lrun = 0.0f;
  v8f acc[NT] = {};
  const int rbase = 8 * (lane >> 4);

  for (int jw = 0; jw < NW; ++jw) {
    const int j0 = jw * 32;

    // ---- cooperative B staging: wh rows j0..j0+31, cols colOff..colOff+BN
    for (int idx = tid; idx < 32 * BN; idx += 256) {
      int kk = idx / BN, c = idx % BN;
      int jr = j0 + kk;
      float v = (c < nValid && jr < N_NODES)
                    ? wh[(size_t)jr * ldw + colOff + c] : 0.0f;
      int t = c >> 4, n = c & 15;
      sBf[(t * 32 + bfrag_lane(kk, n)) * 16 + bfrag_u(kk)] = (__bf16)v;
    }
    __syncthreads();

    // ---- masked leaky scores for this lane's 16 (row, col) slots
    const uint32_t w = maskbits[(size_t)row * NW + jw];
    float p[16];
    float lmax = -INFINITY;
    #pragma unroll
    for (int t = 0; t < 16; ++t) {
      int k   = base + t + ((t >= 8) ? 8 : 0);
      int col = j0 + k;
      float e = -INFINITY;
      if (col < N_NODES && ((w >> k) & 1u))
        e = leaky(f1v + f2s[col]);
      p[t] = e;
      lmax = fmaxf(lmax, e);
    }
    float bmax = fmaxf(lmax, __shfl_xor(lmax, 16));
    float newm = fmaxf(mrun, bmax);
    float sc = (newm == -INFINITY || mrun == -INFINITY) ? 0.0f
                                                        : __expf(mrun - newm);
    float ls = 0.0f;
    #pragma unroll
    for (int t = 0; t < 16; ++t) {
      float pv = (p[t] == -INFINITY) ? 0.0f : __expf(p[t] - newm);
      p[t] = pv;
      ls  += pv;
    }
    ls  += __shfl_xor(ls, 16);
    lrun = lrun * sc + ls;
    mrun = newm;

    // ---- rescale accumulator rows (C layout rows differ from score rows)
    float scr[8];
    #pragma unroll
    for (int v = 0; v < 8; ++v) scr[v] = __shfl(sc, rbase + v);
    #pragma unroll
    for (int t = 0; t < NT; ++t)
      #pragma unroll
      for (int v = 0; v < 8; ++v) acc[t][v] *= scr[v];

    // ---- A fragment = exp-weights (bf16)
    v16bf a;
    #pragma unroll
    for (int t = 0; t < 16; ++t) a[t] = (__bf16)p[t];

    // ---- B fragments straight from LDS (2 x ds_load_b128 each) + WMMA
    #pragma unroll
    for (int t = 0; t < NT; ++t) {
      v16bf b = *((const v16bf*)&sBf[(t * 32 + lane) * 16]);
      acc[t] = wmma_bf16(a, b, acc[t]);
    }
    __syncthreads();
  }

  // ---- epilogue: normalize, + bias, leaky, store
  if (active) {
    float inv = (lrun > 0.0f) ? (1.0f / lrun) : 0.0f;
    float invr[8];
    #pragma unroll
    for (int v = 0; v < 8; ++v) invr[v] = __shfl(inv, rbase + v);

    #pragma unroll
    for (int t = 0; t < NT; ++t) {
      int nn = t * 16 + (lane & 15);
      if (nn < nValid) {
        float bv = bias[colOff + nn];
        #pragma unroll
        for (int v = 0; v < 8; ++v) {
          int gr = i0 + rbase + v;
          out[(size_t)gr * ldo + colOff + nn] = leaky(acc[t][v] * invr[v] + bv);
        }
      }
    }
  }
}

// ---------------------------------------------------------------------------
// Kernel 5: logits = leaky(h2 @ wc.T)   [N,2]
// ---------------------------------------------------------------------------
__global__ void classifier(const float* __restrict__ h2,
                           const float* __restrict__ wc,
                           float* __restrict__ logits)
{
  int n = blockIdx.x * blockDim.x + threadIdx.x;
  if (n >= N_NODES) return;
  for (int q = 0; q < 2; ++q) {
    float s = 0.0f;
    for (int c = 0; c < 50; ++c) s += h2[(size_t)n * 50 + c] * wc[q * 50 + c];
    logits[n * 2 + q] = leaky(s);
  }
}

// ---------------------------------------------------------------------------
// Kernel 6: bias head: t = leaky(logits.T @ ws1.T + bs1) [2,60];
//           bias2 = leaky(t @ ws2.T + bs2) [2]
// ---------------------------------------------------------------------------
__global__ __launch_bounds__(256) void bias_head(
    const float* __restrict__ logits, const float* __restrict__ ws1,
    const float* __restrict__ bs1, const float* __restrict__ ws2,
    const float* __restrict__ bs2, float* __restrict__ bias2)
{
  __shared__ float t[120];
  int tid = threadIdx.x;
  if (tid < 120) {
    int q = tid / 60, s = tid % 60;
    float acc = bs1[s];
    for (int n = 0; n < N_NODES; ++n)
      acc += logits[n * 2 + q] * ws1[(size_t)s * N_NODES + n];
    t[tid] = leaky(acc);
  }
  __syncthreads();
  if (tid < 2) {
    float acc = bs2[0];
    for (int s = 0; s < 60; ++s) acc += t[tid * 60 + s] * ws2[s];
    bias2[tid] = leaky(acc);
  }
}

// ---------------------------------------------------------------------------
// Kernel 7: out = [logits + bias2 broadcast ; bias2]
// ---------------------------------------------------------------------------
__global__ void finalize(const float* __restrict__ logits,
                         const float* __restrict__ bias2,
                         float* __restrict__ out)
{
  int gid = blockIdx.x * blockDim.x + threadIdx.x;
  if (gid < N_NODES * 2) out[gid] = logits[gid] + bias2[gid & 1];
  if (gid < 2) out[N_NODES * 2 + gid] = bias2[gid];
}

// ---------------------------------------------------------------------------
// Host launcher
// ---------------------------------------------------------------------------
extern "C" void kernel_launch(void* const* d_in, const int* in_sizes, int n_in,
                              void* d_out, int out_size, void* d_ws, size_t ws_size,
                              hipStream_t stream) {
  const float* x     = (const float*)d_in[0];
  const float* adj   = (const float*)d_in[1];
  const float* w1    = (const float*)d_in[2];
  const float* asrc1 = (const float*)d_in[3];
  const float* adst1 = (const float*)d_in[4];
  const float* b1    = (const float*)d_in[5];
  const float* w2    = (const float*)d_in[6];
  const float* asrc2 = (const float*)d_in[7];
  const float* adst2 = (const float*)d_in[8];
  const float* b2    = (const float*)d_in[9];
  const float* wc    = (const float*)d_in[10];
  const float* ws1   = (const float*)d_in[11];
  const float* bs1   = (const float*)d_in[12];
  const float* bs2   = (const float*)d_in[13];
  const float* ws2   = (const float*)d_in[14];
  float* out = (float*)d_out;

  // workspace carve-up (256B aligned regions)
  char* wsb = (char*)d_ws;
  auto carve = [&](size_t bytes) -> void* {
    void* p = (void*)wsb;
    wsb += (bytes + 255) & ~(size_t)255;
    return p;
  };
  float*    wh1   = (float*)carve((size_t)N_NODES * 120 * 4);
  float*    h1    = (float*)carve((size_t)N_NODES * 120 * 4);
  float*    wh2   = (float*)carve((size_t)N_NODES * 50 * 4);
  float*    h2    = (float*)carve((size_t)N_NODES * 50 * 4);
  float*    f1a   = (float*)carve((size_t)4 * N_NODES * 4);
  float*    f2a   = (float*)carve((size_t)4 * N_NODES * 4);
  float*    f1b   = (float*)carve((size_t)N_NODES * 4);
  float*    f2b   = (float*)carve((size_t)N_NODES * 4);
  float*    lgt   = (float*)carve((size_t)N_NODES * 2 * 4);
  float*    bias2 = (float*)carve(256);
  uint32_t* mask  = (uint32_t*)carve((size_t)N_NODES * NW * 4);

  // 1. adjacency bitmask (read adj once; 4.5MB result lives in L2)
  build_mask<<<(N_NODES * NWPAD) / 256, 256, 0, stream>>>(adj, mask);

  // 2. wh1 = x @ w1flat   [6000,6000]x[6000,120]  (bmode=1: GAT w1 layout)
  gemm_bf16<8><<<47, 256, 0, stream>>>(x, 6000, w1, 0, 1, wh1, 120,
                                       6000, 120, 6000);

  // 3. f1/f2 for layer 1 (4 heads)
  attn_coeff<<<(4 * N_NODES + 255) / 256, 256, 0, stream>>>(
      wh1, 120, 30, 4, asrc1, adst1, f1a, f2a);

  // 4. fused attention layer 1 -> h1 = leaky(concat-head aggregate + b1)
  attn_fused<2><<<4 * 47, 256, 0, stream>>>(
      mask, wh1, 120, /*colStride=*/30, /*nValid=*/30,
      f1a, f2a, b1, h1, 120, /*blocksPerHead=*/47, /*numRB=*/375);

  // 5. wh2 = h1 @ w2   [6000,120]x[120,50]
  gemm_bf16<4><<<47, 256, 0, stream>>>(h1, 120, w2, 50, 0, wh2, 50,
                                       6000, 50, 120);

  // 6. f1/f2 for layer 2 (1 head)
  attn_coeff<<<(N_NODES + 255) / 256, 256, 0, stream>>>(
      wh2, 50, 50, 1, asrc2, adst2, f1b, f2b);

  // 7. fused attention layer 2 -> h2 = leaky(aggregate + b2)
  attn_fused<4><<<47, 256, 0, stream>>>(
      mask, wh2, 50, /*colStride=*/0, /*nValid=*/50,
      f1b, f2b, b2, h2, 50, /*blocksPerHead=*/47, /*numRB=*/375);

  // 8. classifier logits
  classifier<<<(N_NODES + 255) / 256, 256, 0, stream>>>(h2, wc, lgt);

  // 9. global bias head (single block)
  bias_head<<<1, 256, 0, stream>>>(lgt, ws1, bs1, ws2, bs2, bias2);

  // 10. final output
  finalize<<<(N_NODES * 2 + 255) / 256, 256, 0, stream>>>(lgt, bias2, out);
}
